// TransformerModel_22265110462812
// MI455X (gfx1250) — matrix operand order repaired and checked by reading, not compiled, not run on hardware
//
#include <hip/hip_runtime.h>
#include <hip/hip_bf16.h>

// ---------------------------------------------------------------------------
// Mamba stack (32 layers) + LM head for gfx1250 (MI455X).
// GEMMs: v_wmma_f32_16x16x32_bf16, LDS double-buffered via
// global_load_async_to_lds_b128 (ASYNCcnt), unconditional tiles
// (all dims padded to M%128 / N%64 / K%32 == 0).
// Selective scan: per-channel VALU kernel (sequential in L), v_exp_f32.
// ---------------------------------------------------------------------------

typedef __attribute__((ext_vector_type(16))) __bf16 v16bf;
typedef __attribute__((ext_vector_type(8)))  float  v8f;

#define GPTR(T) T __attribute__((address_space(1)))*

#define NLAYER 32
#define DMODEL 256
#define DINNER 512
#define NSTATE 16
#define RRANK  16
#define KCONV  4
#define VOCAB  4096
#define FFDIM  1024
#define BATCH  32
#define SEQ    256
#define MTOK   (BATCH * SEQ)   // 8192 rows in every GEMM
#define DBCS   64              // dbc row stride (48 padded to 64)
#define DTK    32              // dt-proj K (16 padded to 32)

__device__ inline unsigned short f2bf(float f) {
    union { float f; unsigned int u; } a; a.f = f;
    unsigned int r = a.u + 0x7FFFu + ((a.u >> 16) & 1u);   // round-to-nearest-even
    return (unsigned short)(r >> 16);
}

__device__ inline v16bf frag_ld(const unsigned short* p0, const unsigned short* p1) {
    union { uint4 q[2]; v16bf v; } r;
    r.q[0] = *(const uint4*)p0;   // 8 bf16 (low K half)
    r.q[1] = *(const uint4*)p1;   // 8 bf16 (high K half)
    return r.v;
}

// ---------------------------------------------------------------------------
// f32 -> bf16 bulk convert
// ---------------------------------------------------------------------------
__global__ void cvt_bf16_kernel(const float* __restrict__ src,
                                unsigned short* __restrict__ dst, int n) {
    int idx = blockIdx.x * 256 + threadIdx.x;
    if (idx < n) dst[idx] = f2bf(((const GPTR(float))src)[idx]);
}

// f32 (rows x srcK) -> bf16 (rows x dstK), zero-padding cols srcK..dstK-1
__global__ void cvt_pad_kernel(const float* __restrict__ src,
                               unsigned short* __restrict__ dst,
                               int rows, int srcK, int dstK) {
    int idx = blockIdx.x * 256 + threadIdx.x;
    if (idx >= rows * dstK) return;
    int r = idx / dstK, c = idx - r * dstK;
    dst[idx] = (c < srcK) ? f2bf(((const GPTR(float))src)[r * srcK + c])
                          : (unsigned short)0;
}

// ---------------------------------------------------------------------------
// Embedding gather: x[m, d] = emb[ids[m], d]
// ---------------------------------------------------------------------------
__global__ void embed_kernel(const int* __restrict__ ids,
                             const float* __restrict__ emb,
                             float* __restrict__ x) {
    size_t idx = (size_t)blockIdx.x * 256 + threadIdx.x;     // < MTOK*DMODEL
    size_t m = idx >> 8;
    int    d = (int)(idx & 255);
    ((GPTR(float))x)[idx] = ((const GPTR(float))emb)[(size_t)ids[m] * DMODEL + d];
}

// ---------------------------------------------------------------------------
// LayerNorm over D=256: one wave32 per token, 8 elems/lane, shuffle reduce.
// ---------------------------------------------------------------------------
__global__ __launch_bounds__(256)
void ln_kernel(const float* __restrict__ x,
               const float* __restrict__ w,
               const float* __restrict__ b,
               unsigned short* __restrict__ out) {
    int tok  = blockIdx.x * 8 + (threadIdx.x >> 5);
    int lane = threadIdx.x & 31;
    const float* row = x + (size_t)tok * DMODEL;
    float v[8] __attribute__((aligned(16)));
    *(float4*)(v)     = *(const float4*)(row + lane * 8);
    *(float4*)(v + 4) = *(const float4*)(row + lane * 8 + 4);
    float s = 0.f, s2 = 0.f;
#pragma unroll
    for (int j = 0; j < 8; ++j) { s += v[j]; s2 += v[j] * v[j]; }
#pragma unroll
    for (int msk = 16; msk > 0; msk >>= 1) {
        s  += __shfl_xor(s,  msk, 32);
        s2 += __shfl_xor(s2, msk, 32);
    }
    float mu  = s  * (1.f / DMODEL);
    float var = s2 * (1.f / DMODEL) - mu * mu;
    float rs  = rsqrtf(var + 1e-5f);
    unsigned short o[8] __attribute__((aligned(16)));
#pragma unroll
    for (int j = 0; j < 8; ++j) {
        int d = lane * 8 + j;
        o[j] = f2bf((v[j] - mu) * rs * w[d] + b[d]);
    }
    *(uint4*)(out + (size_t)tok * DMODEL + lane * 8) = *(uint4*)o;
}

// ---------------------------------------------------------------------------
// Tiled WMMA GEMM:  C[M,N] = act(A[M,K] @ W[N,K]^T + bias) (+ residual)
//   Block tile 128x64, 256 threads = 8 waves, wave -> 16 rows x 4 n-tiles.
//   Requires M%128==0, N%64==0, K%32==0 (caller pads) -> no guards.
//   A/B tiles staged with global_load_async_to_lds_b128, LDS double-buffered,
//   pipeline synchronized with s_wait_asynccnt + workgroup barrier.
// ---------------------------------------------------------------------------
#define LDSA_STRIDE 40   // bf16 elems; 80B rows keep 16B alignment, dodge banks
__global__ __launch_bounds__(256)
void wmma_gemm_kernel(const unsigned short* __restrict__ A,
                      const unsigned short* __restrict__ W,
                      const float* __restrict__ bias,
                      const float* __restrict__ resid,
                      float* __restrict__ Cf,
                      unsigned short* __restrict__ Cb,
                      int M, int N, int K, int relu) {
    __shared__ unsigned short As[2][128 * LDSA_STRIDE];
    __shared__ unsigned short Bs[2][64  * LDSA_STRIDE];

    const int tid  = threadIdx.x;
    const int lane = tid & 31;
    const int wave = tid >> 5;
    const int half = lane >> 4;
    const int l16  = lane & 15;
    const int m0   = blockIdx.y * 128;
    const int n0   = blockIdx.x * 64;

    // staging assignment: A row=tid>>1 colseg=(tid&1)*16 (two b128);
    //                     B n-row=tid>>2 kseg=(tid&3)*8 (one b128)
    const int ar  = tid >> 1, acs = (tid & 1) * 16;
    const int bn  = tid >> 2, bks = (tid & 3) * 8;
    const unsigned short* Aip = A + (size_t)(m0 + ar) * K + acs;
    const unsigned short* Wip = W + (size_t)(n0 + bn) * K + bks;
    const unsigned laA[2] = {
        (unsigned)(unsigned long long)&As[0][ar * LDSA_STRIDE + acs],
        (unsigned)(unsigned long long)&As[1][ar * LDSA_STRIDE + acs] };
    const unsigned laB[2] = {
        (unsigned)(unsigned long long)&Bs[0][bn * LDSA_STRIDE + bks],
        (unsigned)(unsigned long long)&Bs[1][bn * LDSA_STRIDE + bks] };

    auto issue = [&](int k0, int buf) {
        unsigned long long ga = (unsigned long long)(Aip + k0);
        // INST_OFFSET applies to both LDS and global address (ISA 15.18.3)
        asm volatile("global_load_async_to_lds_b128 %0, %1, off"
                     :: "v"(laA[buf]), "v"(ga) : "memory");
        asm volatile("global_load_async_to_lds_b128 %0, %1, off offset:16"
                     :: "v"(laA[buf]), "v"(ga) : "memory");
        unsigned long long gb = (unsigned long long)(Wip + k0);
        asm volatile("global_load_async_to_lds_b128 %0, %1, off"
                     :: "v"(laB[buf]), "v"(gb) : "memory");
    };

    const v8f vz = {0.f, 0.f, 0.f, 0.f, 0.f, 0.f, 0.f, 0.f};
    v8f acc[4] = {vz, vz, vz, vz};

    const int nk = K >> 5;
    issue(0, 0);
    for (int i = 0; i < nk; ++i) {
        asm volatile("s_wait_asynccnt 0x0" ::: "memory");  // this wave's DMA done
        __syncthreads();                                    // tile visible to all
        if (i + 1 < nk) issue((i + 1) << 5, (i + 1) & 1);   // prefetch next tile

        const unsigned short* as = As[i & 1];
        const unsigned short* bs = Bs[i & 1];
        // A fragment (ISA 16-bit A 16x32): VGPR0..3 <- K=8h+{0..7},
        //                                  VGPR4..7 <- K=16+8h+{0..7}
        const unsigned short* arow = &as[(wave * 16 + l16) * LDSA_STRIDE + 8 * half];
        v16bf afrag = frag_ld(arow, arow + 16);
#pragma unroll
        for (int nt = 0; nt < 4; ++nt) {
            // B fragment (32x16): same K order as A: VGPR0..3 <- K=8h+{0..7}, VGPR4..7 <- K=16+8h+{0..7}
            const unsigned short* brow = &bs[(nt * 16 + l16) * LDSA_STRIDE + 8 * half];
            v16bf bfrag = frag_ld(brow, brow + 16);
            acc[nt] = __builtin_amdgcn_wmma_f32_16x16x32_bf16(
                false, afrag, false, bfrag, (short)0, acc[nt], false, false);
        }
    }

    // Epilogue: C/D layout — VGPR r holds (m = r + 8*half, n = lane&15)
#pragma unroll
    for (int nt = 0; nt < 4; ++nt) {
        int gcol = n0 + nt * 16 + l16;
        float bv = bias ? ((const GPTR(float))bias)[gcol] : 0.f;
#pragma unroll
        for (int r = 0; r < 8; ++r) {
            int grow = m0 + wave * 16 + r + 8 * half;
            float val = acc[nt][r] + bv;
            if (relu) val = fmaxf(val, 0.f);
            size_t o = (size_t)grow * N + gcol;
            if (resid) val += ((const GPTR(float))resid)[o];
            if (Cf) ((GPTR(float))Cf)[o] = val;
            if (Cb) ((GPTR(unsigned short))Cb)[o] = f2bf(val);
        }
    }
}

// ---------------------------------------------------------------------------
// Causal depthwise conv1d (K=4) + SiLU.  xm = xz[:, :, :DINNER].
// ---------------------------------------------------------------------------
__global__ void conv_silu_kernel(const float* __restrict__ xz,
                                 const float* __restrict__ cw,
                                 const float* __restrict__ cb,
                                 float* __restrict__ xc,
                                 unsigned short* __restrict__ xcb) {
    size_t idx = (size_t)blockIdx.x * 256 + threadIdx.x;  // < MTOK*DINNER
    int    c   = (int)(idx & (DINNER - 1));
    size_t m   = idx >> 9;
    int    l   = (int)(m & (SEQ - 1));
    const GPTR(float) gxz = (const GPTR(float))xz;
    float acc = ((const GPTR(float))cb)[c];
#pragma unroll
    for (int k = 0; k < KCONV; ++k) {
        int lk = l + k - (KCONV - 1);
        if (lk >= 0)
            acc += gxz[(m + k - (KCONV - 1)) * (2 * DINNER) + c]
                   * ((const GPTR(float))cw)[c * KCONV + k];
    }
    float sv = acc / (1.f + __expf(-acc));
    ((GPTR(float))xc)[idx]          = sv;
    ((GPTR(unsigned short))xcb)[idx] = f2bf(sv);
}

// ---------------------------------------------------------------------------
// Pack dt-rank cols of dbc as bf16 A-operand (M x DTK), zero-padded K.
// ---------------------------------------------------------------------------
__global__ void prep_dt_kernel(const float* __restrict__ dbc,
                               unsigned short* __restrict__ dtA) {
    int idx = blockIdx.x * 256 + threadIdx.x;            // < MTOK*DTK
    int r   = idx & (DTK - 1);
    size_t m = (size_t)(idx >> 5);
    ((GPTR(unsigned short))dtA)[idx] =
        (r < RRANK) ? f2bf(((const GPTR(float))dbc)[m * DBCS + r]) : (unsigned short)0;
}

// ---------------------------------------------------------------------------
// Selective scan: block per batch element, thread per inner channel.
// State s[16] in regs; B_t/C_t broadcast via LDS per timestep.
// Fused: softplus(dt), dA = exp(dt*A), Dskip, SiLU(z) gate, bf16 output.
// ---------------------------------------------------------------------------
__global__ __launch_bounds__(512)
void scan_kernel(const float* __restrict__ dbc,
                 const float* __restrict__ dt_raw,
                 const float* __restrict__ xc,
                 const float* __restrict__ xz,
                 const float* __restrict__ A_log,
                 const float* __restrict__ dt_b,
                 const float* __restrict__ Dskip,
                 unsigned short* __restrict__ y_out) {
    __shared__ float sBC[2 * NSTATE];
    const int b = blockIdx.x;
    const int c = threadIdx.x;
    const GPTR(float) gdbc = (const GPTR(float))dbc;
    const GPTR(float) gdt  = (const GPTR(float))dt_raw;
    const GPTR(float) gxc  = (const GPTR(float))xc;
    const GPTR(float) gxz  = (const GPTR(float))xz;

    float a[NSTATE], s[NSTATE];
#pragma unroll
    for (int n = 0; n < NSTATE; ++n) {
        a[n] = -__expf(((const GPTR(float))A_log)[c * NSTATE + n]);
        s[n] = 0.f;
    }
    const float dtb = ((const GPTR(float))dt_b)[c];
    const float Dc  = ((const GPTR(float))Dskip)[c];

    for (int t = 0; t < SEQ; ++t) {
        size_t m = (size_t)b * SEQ + t;
        __syncthreads();
        if (c < 2 * NSTATE)
            sBC[c] = gdbc[m * DBCS + RRANK + c];
        __syncthreads();

        float dv = gdt[m * DINNER + c] + dtb;
        dv = (dv > 15.f) ? dv : __logf(1.f + __expf(dv));   // softplus
        float xt = gxc[m * DINNER + c];
        float sc = dv * xt;
        float y  = 0.f;
#pragma unroll
        for (int n = 0; n < NSTATE; ++n) {
            float dA = __expf(dv * a[n]);
            s[n] = s[n] * dA + sc * sBC[n];
            y += s[n] * sBC[NSTATE + n];
        }
        float z = gxz[m * (2 * DINNER) + DINNER + c];
        float g = z / (1.f + __expf(-z));                   // SiLU gate
        ((GPTR(unsigned short))y_out)[m * DINNER + c] = f2bf((y + xt * Dc) * g);
    }
}

// ---------------------------------------------------------------------------
// Host-side orchestration
// ---------------------------------------------------------------------------
extern "C" void kernel_launch(void* const* d_in, const int* in_sizes, int n_in,
                              void* d_out, int out_size, void* d_ws, size_t ws_size,
                              hipStream_t stream) {
    (void)in_sizes; (void)n_in; (void)out_size; (void)ws_size;

    const int*   ids     = (const int*)  d_in[0];
    const float* emb     = (const float*)d_in[1];
    const float* ln_w    = (const float*)d_in[2];
    const float* ln_b    = (const float*)d_in[3];
    const float* in_w    = (const float*)d_in[4];
    const float* conv_w  = (const float*)d_in[5];
    const float* conv_b  = (const float*)d_in[6];
    const float* xp_w    = (const float*)d_in[7];
    const float* dt_w    = (const float*)d_in[8];
    const float* dt_b    = (const float*)d_in[9];
    const float* A_log   = (const float*)d_in[10];
    const float* Dskip   = (const float*)d_in[11];
    const float* out_w   = (const float*)d_in[12];
    const float* W1      = (const float*)d_in[13];
    const float* b1      = (const float*)d_in[14];
    const float* W2      = (const float*)d_in[15];
    const float* b2      = (const float*)d_in[16];

    // ---- workspace partition ----
    char* p = (char*)d_ws;
    auto alloc = [&](size_t bytes) -> char* {
        char* r = p; p += (bytes + 255) & ~(size_t)255; return r;
    };
    const size_t M = MTOK;
    unsigned short* wb_in  = (unsigned short*)alloc((size_t)NLAYER * 2 * DINNER * DMODEL * 2);
    unsigned short* wb_xp  = (unsigned short*)alloc((size_t)NLAYER * DBCS * DINNER * 2);   // 48->64 rows (pad rows unread)
    unsigned short* wb_dt  = (unsigned short*)alloc((size_t)NLAYER * DINNER * DTK * 2);    // K 16->32 zero-pad
    unsigned short* wb_out = (unsigned short*)alloc((size_t)NLAYER * DMODEL * DINNER * 2);
    unsigned short* wb_W1  = (unsigned short*)alloc((size_t)FFDIM * DMODEL * 2);
    unsigned short* wb_W2  = (unsigned short*)alloc((size_t)VOCAB * FFDIM * 2);
    float*          xf     = (float*)         alloc(M * DMODEL * 4);
    unsigned short* h16    = (unsigned short*)alloc(M * DMODEL * 2);
    float*          xzf    = (float*)         alloc(M * 2 * DINNER * 4);
    float*          xcf    = (float*)         alloc(M * DINNER * 4);
    unsigned short* xc16   = (unsigned short*)alloc(M * DINNER * 2);
    float*          dbc    = (float*)         alloc(M * DBCS * 4);
    unsigned short* dtA16  = (unsigned short*)alloc(M * DTK * 2);
    float*          dtr    = (float*)         alloc(M * DINNER * 4);
    unsigned short* y16    = (unsigned short*)alloc(M * DINNER * 2);
    unsigned short* hff16  = (unsigned short*)alloc(M * FFDIM * 2);

    auto cvt = [&](const float* s, unsigned short* d, int n) {
        cvt_bf16_kernel<<<(n + 255) / 256, 256, 0, stream>>>(s, d, n);
    };
    // ---- weights -> bf16 (deterministic every call) ----
    cvt(in_w,  wb_in,  NLAYER * 2 * DINNER * DMODEL);
    for (int l = 0; l < NLAYER; ++l)   // x_proj rows 0..47 of each 64-row slab
        cvt(xp_w + (size_t)l * 48 * DINNER, wb_xp + (size_t)l * DBCS * DINNER, 48 * DINNER);
    cvt_pad_kernel<<<(NLAYER * DINNER * DTK + 255) / 256, 256, 0, stream>>>(
        dt_w, wb_dt, NLAYER * DINNER, RRANK, DTK);
    cvt(out_w, wb_out, NLAYER * DMODEL * DINNER);
    cvt(W1,    wb_W1,  FFDIM * DMODEL);
    cvt(W2,    wb_W2,  VOCAB * FFDIM);

    // ---- embedding ----
    embed_kernel<<<(int)(M * DMODEL / 256), 256, 0, stream>>>(ids, emb, xf);

    const int GY = MTOK / 128;   // 64
    // ---- 32 Mamba layers ----
    for (int l = 0; l < NLAYER; ++l) {
        ln_kernel<<<MTOK / 8, 256, 0, stream>>>(xf, ln_w + l * DMODEL, ln_b + l * DMODEL, h16);

        // in_proj: (M,256) x (1024,256)^T -> xz (M,1024)
        wmma_gemm_kernel<<<dim3((2 * DINNER) / 64, GY), 256, 0, stream>>>(
            h16, wb_in + (size_t)l * 2 * DINNER * DMODEL,
            nullptr, nullptr, xzf, nullptr, MTOK, 2 * DINNER, DMODEL, 0);

        conv_silu_kernel<<<(int)(M * DINNER / 256), 256, 0, stream>>>(
            xzf, conv_w + (size_t)l * DINNER * KCONV, conv_b + (size_t)l * DINNER, xcf, xc16);

        // x_proj: (M,512) x (64,512)^T -> dbc (M,64), cols 48..63 unread
        wmma_gemm_kernel<<<dim3(DBCS / 64, GY), 256, 0, stream>>>(
            xc16, wb_xp + (size_t)l * DBCS * DINNER,
            nullptr, nullptr, dbc, nullptr, MTOK, DBCS, DINNER, 0);

        prep_dt_kernel<<<(int)(M * DTK / 256), 256, 0, stream>>>(dbc, dtA16);

        // dt proj: (M,32) x (512,32)^T -> dt_raw (M,512)
        wmma_gemm_kernel<<<dim3(DINNER / 64, GY), 256, 0, stream>>>(
            dtA16, wb_dt + (size_t)l * DINNER * DTK,
            nullptr, nullptr, dtr, nullptr, MTOK, DINNER, DTK, 0);

        scan_kernel<<<BATCH, DINNER, 0, stream>>>(
            dbc, dtr, xcf, xzf,
            A_log + (size_t)l * DINNER * NSTATE,
            dt_b + (size_t)l * DINNER,
            Dskip + (size_t)l * DINNER, y16);

        // out_proj + residual (in place on xf): (M,512) x (256,512)^T
        wmma_gemm_kernel<<<dim3(DMODEL / 64, GY), 256, 0, stream>>>(
            y16, wb_out + (size_t)l * DMODEL * DINNER,
            nullptr, xf, xf, nullptr, MTOK, DMODEL, DINNER, 0);
    }

    // ---- LM head ----
    cvt(xf, h16, (int)(M * DMODEL));
    // FF1: relu(x @ W1^T + b1) -> bf16
    wmma_gemm_kernel<<<dim3(FFDIM / 64, GY), 256, 0, stream>>>(
        h16, wb_W1, b1, nullptr, nullptr, hff16, MTOK, FFDIM, DMODEL, 1);
    // FF2: h @ W2^T + b2 -> f32 logits
    wmma_gemm_kernel<<<dim3(VOCAB / 64, GY), 256, 0, stream>>>(
        hff16, wb_W2, b2, nullptr, (float*)d_out, nullptr, MTOK, VOCAB, FFDIM, 0);
}
